// OdeintSolver_65962107732831
// MI455X (gfx1250) — matrix operand order repaired and checked by reading, not compiled, not run on hardware
//
#include <hip/hip_runtime.h>
#include <hip/hip_bf16.h>

#define BB 256   // batch
#define NN 512   // state dim
#define HH 2048  // hidden dim
#define TT 64    // time points
#define DTF 0.05f

typedef __attribute__((ext_vector_type(16))) __bf16 v16bf;
typedef __attribute__((ext_vector_type(8)))  float  v8f;

union FragBF {
    v16bf  v;
    uint4  q[2];
    __bf16 h[16];
};

// ---------------------------------------------------------------------------
// CDNA5 async global->LDS copy (ASYNCcnt path), 16 bytes per lane.
// vdst = LDS byte address (low 32 bits of flat shared pointer), vaddr = 64-bit
// global address, saddr = off  (GV mode).
// ---------------------------------------------------------------------------
__device__ __forceinline__ void async_b128(const void* gp, void* lp) {
    unsigned lds = (unsigned)(unsigned long long)(uintptr_t)lp;
    asm volatile("global_load_async_to_lds_b128 %0, %1, off"
                 :: "v"(lds), "v"(gp)
                 : "memory");
}
__device__ __forceinline__ void wait_async0() {
    asm volatile("s_wait_asynccnt 0" ::: "memory");
}

// ---------------------------------------------------------------------------
// Weight prep: cast fp32 weights to bf16, transposed so that the GEMM B-matrix
// (columns of W) is row-contiguous:  W1T[h][n] = W1[n][h],  W2T[n][h] = W2[h][n]
// ---------------------------------------------------------------------------
__global__ __launch_bounds__(256) void prep_w1(const float* __restrict__ w1,
                                               __bf16* __restrict__ w1t) {
    int idx = blockIdx.x * 256 + threadIdx.x;
    if (idx >= NN * HH) return;
    int h = idx / NN;
    int n = idx - h * NN;
    w1t[idx] = (__bf16)w1[(size_t)n * HH + h];
}

__global__ __launch_bounds__(256) void prep_w2(const float* __restrict__ w2,
                                               __bf16* __restrict__ w2t) {
    int idx = blockIdx.x * 256 + threadIdx.x;
    if (idx >= NN * HH) return;
    int n = idx / HH;
    int h = idx - n * HH;
    w2t[idx] = (__bf16)w2[(size_t)h * NN + n];
}

// ---------------------------------------------------------------------------
// Init: ycur = y0, y_t[:,0,:] = y0, t_t = t_eval, y_loss = 0
// ---------------------------------------------------------------------------
__global__ __launch_bounds__(256) void init_state(const float* __restrict__ y0,
                                                  const float* __restrict__ te,
                                                  float* __restrict__ ycur,
                                                  float* __restrict__ yt,
                                                  float* __restrict__ tt,
                                                  float* __restrict__ yloss) {
    int idx = blockIdx.x * 256 + threadIdx.x;
    if (idx < BB * NN) {
        float v = y0[idx];
        ycur[idx] = v;
        int b = idx / NN;
        int n = idx - b * NN;
        yt[(size_t)b * TT * NN + n] = v;   // t = 0 slice
    }
    if (idx < TT) tt[idx] = te[idx];
    if (idx < BB) yloss[idx] = 0.0f;
}

// ---------------------------------------------------------------------------
// Tile staging for one K-slice (32) of a 128x128 output tile.
//   MODE 0: A is f32 -> VALU convert to bf16 + ds_store (cvt_pk_bf16_f32)
//   MODE 1: A is bf16 -> async global->LDS copy
//   B (bf16 transposed weights) always goes through the async path.
// Per thread: one (row, 16-half chunk): srow = tid>>1, scol = (tid&1)*16.
// ---------------------------------------------------------------------------
template <int KD, int MODE>
__device__ __forceinline__ void stage_tiles(const void* Aptr,
                                            const __bf16* BT,
                                            __bf16* AsB, __bf16* BsB,
                                            int tileM, int tileN, int k0,
                                            int srow, int scol) {
    if (MODE == 0) {
        const float* A = (const float*)Aptr;
        const float4* src =
            (const float4*)(A + (size_t)(tileM + srow) * KD + (k0 + scol));
        float4 f0 = src[0], f1 = src[1], f2 = src[2], f3 = src[3];
        FragBF fa;
        fa.h[0]  = (__bf16)f0.x;  fa.h[1]  = (__bf16)f0.y;
        fa.h[2]  = (__bf16)f0.z;  fa.h[3]  = (__bf16)f0.w;
        fa.h[4]  = (__bf16)f1.x;  fa.h[5]  = (__bf16)f1.y;
        fa.h[6]  = (__bf16)f1.z;  fa.h[7]  = (__bf16)f1.w;
        fa.h[8]  = (__bf16)f2.x;  fa.h[9]  = (__bf16)f2.y;
        fa.h[10] = (__bf16)f2.z;  fa.h[11] = (__bf16)f2.w;
        fa.h[12] = (__bf16)f3.x;  fa.h[13] = (__bf16)f3.y;
        fa.h[14] = (__bf16)f3.z;  fa.h[15] = (__bf16)f3.w;
        uint4* dst = (uint4*)&AsB[srow * 32 + scol];
        dst[0] = fa.q[0];
        dst[1] = fa.q[1];
    } else {
        const __bf16* A = (const __bf16*)Aptr;
        const __bf16* g = A + (size_t)(tileM + srow) * KD + (k0 + scol);
        __bf16* l = &AsB[srow * 32 + scol];
        async_b128(g, l);
        async_b128(g + 8, l + 8);
    }
    {
        const __bf16* g = BT + (size_t)(tileN + srow) * KD + (k0 + scol);
        __bf16* l = &BsB[srow * 32 + scol];
        async_b128(g, l);
        async_b128(g + 8, l + 8);
    }
}

// ---------------------------------------------------------------------------
// Fused WMMA GEMM, double-buffered LDS with async fills.
//   MODE 0:  H = tanh(A_f32[M,KD] @ W^T + bias)        -> hout (bf16)
//   MODE 1:  k = A_bf16[M,KD] @ W^T + bias, RK4 epilogue per `stage`
// Tile: 128x128 per workgroup (8 waves); wave computes 64x32 = 4x2 WMMA tiles.
// ---------------------------------------------------------------------------
template <int KD, int NO, int MODE>
__global__ __launch_bounds__(256, 2)
void wmma_gemm(const void* __restrict__ Aptr,
               const __bf16* __restrict__ BT,     // [NO][KD] bf16 (transposed W)
               const float* __restrict__ bias,    // [NO]
               __bf16* __restrict__ hout,         // MODE 0 output
               float* __restrict__ ycur,          // MODE 1: y at step start (stage4: updated)
               float* __restrict__ accb,          // MODE 1: k1 + 2k2 + 2k3 (+k4)
               float* __restrict__ ytmp,          // MODE 1: next stage input y + c*k
               float* __restrict__ yt,            // MODE 1 stage 4: y_t base [B,T,N]
               int tstep, float c_stage, float w_stage, int stage)
{
    __shared__ __align__(16) __bf16 As[2][128 * 32];
    __shared__ __align__(16) __bf16 Bs[2][128 * 32];

    const int tid   = threadIdx.x;
    const int lane  = tid & 31;               // wave32
    const int wid   = tid >> 5;               // 8 waves
    const int tileM = blockIdx.y * 128;
    const int tileN = blockIdx.x * 128;
    const int waveM = (wid >> 2) * 64;        // 0 / 64
    const int waveN = (wid & 3) * 32;         // 0 / 32 / 64 / 96
    const int lh    = lane & 15;
    const int koffA = (lane < 16) ? 0 : 16;   // A frag: K {koffA..+15} contiguous
    const int koffB = (lane < 16) ? 0 : 16;   // B frag: K {koffB..+15} contiguous

    const int srow = tid >> 1;                // 0..127 staging row
    const int scol = (tid & 1) * 16;          // 0 / 16 halves

    v8f zero = {0.f, 0.f, 0.f, 0.f, 0.f, 0.f, 0.f, 0.f};
    v8f acc[4][2];
#pragma unroll
    for (int i = 0; i < 4; ++i)
#pragma unroll
        for (int j = 0; j < 2; ++j) acc[i][j] = zero;

    // preload tile 0 into buffer 0
    stage_tiles<KD, MODE>(Aptr, BT, As[0], Bs[0], tileM, tileN, 0, srow, scol);
    wait_async0();
    __syncthreads();

    int buf = 0;
    for (int k0 = 0; k0 < KD; k0 += 32) {
        const int nxt = buf ^ 1;
        if (k0 + 32 < KD) {
            // fill the other buffer while this one is being consumed
            stage_tiles<KD, MODE>(Aptr, BT, As[nxt], Bs[nxt], tileM, tileN,
                                  k0 + 32, srow, scol);
            if (k0 + 64 < KD)   // pull the tile after next toward the caches
                __builtin_prefetch(
                    (const void*)(BT + (size_t)(tileN + srow) * KD + (k0 + 64 + scol)),
                    0, 3);
        }

        // B fragments: lane holds column N=lh, K koffB..koffB+15 (contiguous 32B)
        FragBF bfr[2];
#pragma unroll
        for (int j = 0; j < 2; ++j) {
            const uint4* p =
                (const uint4*)&Bs[buf][(waveN + 16 * j + lh) * 32 + koffB];
            bfr[j].q[0] = p[0];
            bfr[j].q[1] = p[1];
        }
        // A fragments + 8 WMMAs
#pragma unroll
        for (int i = 0; i < 4; ++i) {
            FragBF af;
            const __bf16* pa = &As[buf][(waveM + 16 * i + lh) * 32 + koffA];
            af.q[0] = *(const uint4*)pa;
            af.q[1] = *(const uint4*)(pa + 8);
#pragma unroll
            for (int j = 0; j < 2; ++j) {
                acc[i][j] = __builtin_amdgcn_wmma_f32_16x16x32_bf16(
                    false, af.v, false, bfr[j].v, (short)0, acc[i][j], false, false);
            }
        }

        wait_async0();     // async fills of `nxt` complete (barrier doesn't cover ASYNCcnt)
        __syncthreads();   // everyone done reading `buf`, ds_stores drained
        buf = nxt;
    }

    // ---- epilogue: C/D layout VGPR r -> M = r (lanes<16) or r+8 (lanes>=16) ----
#pragma unroll
    for (int i = 0; i < 4; ++i) {
#pragma unroll
        for (int j = 0; j < 2; ++j) {
            const int col = tileN + waveN + 16 * j + lh;
            const float bv = bias[col];
            const int rowBase = tileM + waveM + 16 * i + ((lane < 16) ? 0 : 8);
#pragma unroll
            for (int r = 0; r < 8; ++r) {
                const int row = rowBase + r;
                const float val = acc[i][j][r] + bv;
                if (MODE == 0) {
                    hout[(size_t)row * NO + col] = (__bf16)tanhf(val);
                } else {
                    const size_t idx = (size_t)row * NO + col;
                    if (stage == 1) {
                        accb[idx] = val;                         // acc = k1
                        ytmp[idx] = ycur[idx] + c_stage * val;   // y + h/2*k1
                    } else if (stage != 4) {
                        accb[idx] += w_stage * val;              // acc += 2*k
                        ytmp[idx] = ycur[idx] + c_stage * val;
                    } else {
                        const float a  = accb[idx] + val;        // + k4
                        const float yn = ycur[idx] + (DTF / 6.0f) * a;
                        ycur[idx] = yn;                          // next step's y
                        yt[((size_t)row * TT + tstep) * NN + col] = yn;
                    }
                }
            }
        }
    }
}

// ---------------------------------------------------------------------------
// Host orchestration: 2 prep + 1 init + 63 steps x (4 stages x 2 GEMMs)
// ---------------------------------------------------------------------------
extern "C" void kernel_launch(void* const* d_in, const int* in_sizes, int n_in,
                              void* d_out, int out_size, void* d_ws, size_t ws_size,
                              hipStream_t stream) {
    const float* y0 = (const float*)d_in[0];
    const float* te = (const float*)d_in[1];
    const float* W1 = (const float*)d_in[2];
    const float* b1 = (const float*)d_in[3];
    const float* W2 = (const float*)d_in[4];
    const float* b2 = (const float*)d_in[5];

    float* yt  = (float*)d_out;                  // [B,T,N]
    float* tt  = yt + (size_t)BB * TT * NN;      // [T]
    float* yls = tt + TT;                        // [B]

    char*  ws  = (char*)d_ws;
    size_t off = 0;
    auto carve = [&](size_t bytes) {
        char* p = ws + off;
        off = (off + bytes + 255) & ~(size_t)255;
        return p;
    };
    __bf16* W1T  = (__bf16*)carve((size_t)HH * NN * 2);  // [H][N]
    __bf16* W2T  = (__bf16*)carve((size_t)NN * HH * 2);  // [N][H]
    __bf16* hbf  = (__bf16*)carve((size_t)BB * HH * 2);  // hidden (bf16)
    float*  ycur = (float*)carve((size_t)BB * NN * 4);
    float*  ytmp = (float*)carve((size_t)BB * NN * 4);
    float*  accb = (float*)carve((size_t)BB * NN * 4);

    prep_w1<<<(NN * HH + 255) / 256, 256, 0, stream>>>(W1, W1T);
    prep_w2<<<(NN * HH + 255) / 256, 256, 0, stream>>>(W2, W2T);
    init_state<<<(BB * NN + 255) / 256, 256, 0, stream>>>(y0, te, ycur, yt, tt, yls);

    const dim3 g1(HH / 128, BB / 128);  // (16, 2)
    const dim3 g2(NN / 128, BB / 128);  // (4, 2)

    for (int t = 1; t < TT; ++t) {
        // stage 1: k1 = f(y)
        wmma_gemm<NN, HH, 0><<<g1, 256, 0, stream>>>(ycur, W1T, b1, hbf,
            nullptr, nullptr, nullptr, nullptr, 0, 0.f, 0.f, 0);
        wmma_gemm<HH, NN, 1><<<g2, 256, 0, stream>>>(hbf, W2T, b2, nullptr,
            ycur, accb, ytmp, yt, t, 0.5f * DTF, 1.f, 1);
        // stage 2: k2 = f(y + h/2 k1)
        wmma_gemm<NN, HH, 0><<<g1, 256, 0, stream>>>(ytmp, W1T, b1, hbf,
            nullptr, nullptr, nullptr, nullptr, 0, 0.f, 0.f, 0);
        wmma_gemm<HH, NN, 1><<<g2, 256, 0, stream>>>(hbf, W2T, b2, nullptr,
            ycur, accb, ytmp, yt, t, 0.5f * DTF, 2.f, 2);
        // stage 3: k3 = f(y + h/2 k2)
        wmma_gemm<NN, HH, 0><<<g1, 256, 0, stream>>>(ytmp, W1T, b1, hbf,
            nullptr, nullptr, nullptr, nullptr, 0, 0.f, 0.f, 0);
        wmma_gemm<HH, NN, 1><<<g2, 256, 0, stream>>>(hbf, W2T, b2, nullptr,
            ycur, accb, ytmp, yt, t, DTF, 2.f, 3);
        // stage 4: k4 = f(y + h k3); y += h/6 * acc; write y_t[:,t,:]
        wmma_gemm<NN, HH, 0><<<g1, 256, 0, stream>>>(ytmp, W1T, b1, hbf,
            nullptr, nullptr, nullptr, nullptr, 0, 0.f, 0.f, 0);
        wmma_gemm<HH, NN, 1><<<g2, 256, 0, stream>>>(hbf, W2T, b2, nullptr,
            ycur, accb, ytmp, yt, t, 0.f, 1.f, 4);
    }

    (void)in_sizes; (void)n_in; (void)out_size; (void)ws_size;
}
